// FastSA_Layer_15032385536571
// MI455X (gfx1250) — matrix operand order repaired and checked by reading, not compiled, not run on hardware
//
#include <hip/hip_runtime.h>
#include <hip/hip_bf16.h>

// ---------------------------------------------------------------------------
// MI455X (gfx1250, wave32) offset-attention layer.
// v_wmma_f32_16x16x32_bf16 everywhere; flash attention (no 256MB scores);
// all fragments are contiguous b128 loads; double-buffered LDS staging.
// ---------------------------------------------------------------------------

#define BDIM 16
#define CDIM 256
#define NDIM 2048

typedef __attribute__((ext_vector_type(16))) __bf16 v16bf;
typedef __attribute__((ext_vector_type(8)))  __bf16 v8bf;
typedef __attribute__((ext_vector_type(8)))  float  v8f;

union bf16x16 { v16bf v; v8bf h[2]; };
union bf16x4  { __bf16 b[4]; uint2 u2; };
union bf16x2  { __bf16 b[2]; unsigned int u; };

__device__ __forceinline__ v8f v8f_zero() {
  v8f z;
#pragma unroll
  for (int i = 0; i < 8; ++i) z[i] = 0.0f;
  return z;
}

__device__ __forceinline__ v8f wmma_bf16(v16bf a, v16bf b, v8f c) {
  return __builtin_amdgcn_wmma_f32_16x16x32_bf16(false, a, false, b,
                                                 (short)0, c, false, false);
}

// A-fragment (16x32 bf16 MxK), row-major source [M][ldk]:
// lane (m,h) elements = {K = 16h+0..15} -> 2x b128.
__device__ __forceinline__ v16bf load_a_frag_rm(const __bf16* base, int ldk) {
  int lane = threadIdx.x & 31;
  int m = lane & 15, h = lane >> 4;
  const __bf16* row = base + (size_t)m * ldk + 16 * h;
  bf16x16 u;
  u.h[0] = *(const v8bf*)(row);
  u.h[1] = *(const v8bf*)(row + 8);
  return u.v;
}

// B-fragment (32x16 bf16 KxN), n-major source [N][ldk]:
// lane (n,h) elements = {K = 16h+0..15} -> 32B contiguous (2x b128).
__device__ __forceinline__ v16bf load_b_frag_nk(const __bf16* base, int ldk) {
  int lane = threadIdx.x & 31;
  int n = lane & 15, h = lane >> 4;
  const __bf16* p = base + (size_t)n * ldk + 16 * h;
  bf16x16 u;
  u.h[0] = *(const v8bf*)(p);
  u.h[1] = *(const v8bf*)(p + 8);
  return u.v;
}

// ---------------------------------------------------------------------------
// Kernel 1: qkv = x^T @ w_qkv^T. q,k stored [B,N,C]; V stored TRANSPOSED
// ([B,C,N]) so attention's P*V B-fragments are contiguous.
// ---------------------------------------------------------------------------
__global__ void __launch_bounds__(128)
qkv_kernel(const float* __restrict__ x, const float* __restrict__ w,
           __bf16* __restrict__ q, __bf16* __restrict__ kk,
           __bf16* __restrict__ vt) {
  __shared__ alignas(16) __bf16 lA[2][64][32];  // [m=n][k=c]
  __shared__ alignas(16) __bf16 lB[2][64][32];  // [d][k=c]
  int b = blockIdx.z, n0 = blockIdx.x * 64, d0 = blockIdx.y * 64;
  int tid = threadIdx.x, lane = tid & 31, wid = tid >> 5;
  int wm = (wid & 1) * 32, wn = (wid >> 1) * 32;

  auto stage = [&](int buf, int kc) {
    // A: two adjacent c-rows of x -> packed (k,k+1) b32 stores, 4 m per item.
#pragma unroll
    for (int i = 0; i < 2; ++i) {
      int l = tid + i * 128;           // 256 items
      int mg = l & 15, kp = l >> 4;    // 16 m-groups x 16 c-pairs
      const float* p0 =
          x + ((size_t)b * CDIM + kc + 2 * kp) * NDIM + n0 + mg * 4;
      float4 f0 = *(const float4*)(p0);
      float4 f1 = *(const float4*)(p0 + NDIM);
      bf16x2 pk;
      pk.b[0] = (__bf16)f0.x; pk.b[1] = (__bf16)f1.x;
      *(unsigned int*)&lA[buf][mg * 4 + 0][2 * kp] = pk.u;
      pk.b[0] = (__bf16)f0.y; pk.b[1] = (__bf16)f1.y;
      *(unsigned int*)&lA[buf][mg * 4 + 1][2 * kp] = pk.u;
      pk.b[0] = (__bf16)f0.z; pk.b[1] = (__bf16)f1.z;
      *(unsigned int*)&lA[buf][mg * 4 + 2][2 * kp] = pk.u;
      pk.b[0] = (__bf16)f0.w; pk.b[1] = (__bf16)f1.w;
      *(unsigned int*)&lA[buf][mg * 4 + 3][2 * kp] = pk.u;
    }
#pragma unroll
    for (int i = 0; i < 4; ++i) {
      int l = tid + i * 128;
      int dl = l >> 3, kp = l & 7;
      float4 wv = *(const float4*)(w + (size_t)(d0 + dl) * CDIM + kc + kp * 4);
      bf16x4 pk;
      pk.b[0] = (__bf16)wv.x; pk.b[1] = (__bf16)wv.y;
      pk.b[2] = (__bf16)wv.z; pk.b[3] = (__bf16)wv.w;
      *(uint2*)&lB[buf][dl][kp * 4] = pk.u2;
    }
  };

  v8f acc[2][2];
#pragma unroll
  for (int i = 0; i < 2; ++i)
#pragma unroll
    for (int j = 0; j < 2; ++j) acc[i][j] = v8f_zero();

  stage(0, 0);
  __syncthreads();
#pragma unroll
  for (int step = 0; step < 8; ++step) {
    int cur = step & 1;
    if (step < 7) stage(cur ^ 1, (step + 1) * 32);
#pragma unroll
    for (int i = 0; i < 2; ++i) {
      v16bf a = load_a_frag_rm(&lA[cur][wm + i * 16][0], 32);
#pragma unroll
      for (int j = 0; j < 2; ++j) {
        v16bf bb = load_b_frag_nk(&lB[cur][wn + j * 16][0], 32);
        acc[i][j] = wmma_bf16(a, bb, acc[i][j]);
      }
    }
    __syncthreads();
  }

  int h = lane >> 4, col = lane & 15;
#pragma unroll
  for (int i = 0; i < 2; ++i)
#pragma unroll
    for (int j = 0; j < 2; ++j) {
      int dcol = d0 + wn + j * 16 + col;   // uniform q/k/v branch per wave
      if (dcol < 512) {
        __bf16* dst = (dcol < 256) ? q : kk;
        int dc = dcol & 255;
#pragma unroll
        for (int r = 0; r < 8; ++r) {
          int nr = n0 + wm + i * 16 + r + 8 * h;
          dst[((size_t)b * NDIM + nr) * CDIM + dc] = (__bf16)acc[i][j][r];
        }
      } else {
        int dc = dcol - 512;  // V transposed: 8 contiguous n -> one b128
        v8bf pv;
#pragma unroll
        for (int r = 0; r < 8; ++r) pv[r] = (__bf16)acc[i][j][r];
        *(v8bf*)(vt + ((size_t)b * CDIM + dc) * NDIM + n0 + wm + i * 16 +
                 8 * h) = pv;
      }
    }
}

// ---------------------------------------------------------------------------
// Kernel 2: flash attention, double-buffered KV tiles (68KB LDS).
// ---------------------------------------------------------------------------
__global__ void __launch_bounds__(128)
attn_kernel(const __bf16* __restrict__ q, const __bf16* __restrict__ kten,
            const __bf16* __restrict__ vt, __bf16* __restrict__ xr) {
  __shared__ alignas(16) __bf16 lK[2][32][CDIM];   // [kv][c]
  __shared__ alignas(16) __bf16 lVT[2][CDIM][32];  // [c][kv]
  __shared__ alignas(16) __bf16 lP[4][16][32];     // per-wave P staging
  int b = blockIdx.y;
  int tid = threadIdx.x, lane = tid & 31, wid = tid >> 5;
  int n0 = blockIdx.x * 64 + wid * 16;
  int h = lane >> 4, col = lane & 15;

  auto stageKV = [&](int buf, int kv0) {
    const uint4* gk = (const uint4*)(kten + ((size_t)b * NDIM + kv0) * CDIM);
    uint4* sk = (uint4*)&lK[buf][0][0];
    uint4* sv = (uint4*)&lVT[buf][0][0];
#pragma unroll
    for (int i = 0; i < 8; ++i) {
      int l = tid + i * 128;
      sk[l] = gk[l];
      int c = l >> 2, part = l & 3;
      sv[l] = *(const uint4*)(vt + ((size_t)b * CDIM + c) * NDIM + kv0 +
                              part * 8);
    }
  };

  v16bf aq[8];
  const __bf16* qrow = q + ((size_t)b * NDIM + n0) * CDIM;
#pragma unroll
  for (int cc = 0; cc < 8; ++cc) aq[cc] = load_a_frag_rm(qrow + cc * 32, CDIM);

  v8f o[16];
#pragma unroll
  for (int ct = 0; ct < 16; ++ct) o[ct] = v8f_zero();
  float rmax[8], rsum[8];
#pragma unroll
  for (int r = 0; r < 8; ++r) { rmax[r] = -1e30f; rsum[r] = 0.0f; }
  const float scale = 0.0625f;  // 1/sqrt(256)

  stageKV(0, 0);
  __syncthreads();
  for (int it = 0; it < NDIM / 32; ++it) {
    int cur = it & 1;
    int kv0 = it * 32;
    if (it < NDIM / 32 - 1) stageKV(cur ^ 1, kv0 + 32);
    if (kv0 + 64 < NDIM)
      __builtin_prefetch(kten + ((size_t)b * NDIM + kv0 + 64) * CDIM +
                             tid * 64, 0, 1);

    // S = Q*K^T: two 16x16 column tiles, K-dim 256 in 8 WMMA steps each.
    v8f s0 = v8f_zero(), s1 = v8f_zero();
#pragma unroll
    for (int cc = 0; cc < 8; ++cc) {
      v16bf bk0 = load_b_frag_nk(&lK[cur][0][cc * 32], CDIM);
      v16bf bk1 = load_b_frag_nk(&lK[cur][16][cc * 32], CDIM);
      s0 = wmma_bf16(aq[cc], bk0, s0);
      s1 = wmma_bf16(aq[cc], bk1, s1);
    }
    s0 = s0 * scale;
    s1 = s1 * scale;

    // Online softmax; row reduce across the 16 lanes of each half.
    float alpha[8];
#pragma unroll
    for (int r = 0; r < 8; ++r) {
      float m = fmaxf(s0[r], s1[r]);
#pragma unroll
      for (int msk = 1; msk < 16; msk <<= 1)
        m = fmaxf(m, __shfl_xor(m, msk, 32));
      float nmax = fmaxf(rmax[r], m);
      alpha[r] = __expf(rmax[r] - nmax);
      rmax[r] = nmax;
      float p0 = __expf(s0[r] - nmax);
      float p1 = __expf(s1[r] - nmax);
      float ls = p0 + p1;
#pragma unroll
      for (int msk = 1; msk < 16; msk <<= 1)
        ls += __shfl_xor(ls, msk, 32);
      rsum[r] = rsum[r] * alpha[r] + ls;
      lP[wid][r + 8 * h][col]      = (__bf16)p0;  // D->A transpose staging
      lP[wid][r + 8 * h][16 + col] = (__bf16)p1;  // (per-wave, in-order DS)
    }

    // Vectorized rescale (accumulator VGPR index == row index).
    v8f av;
#pragma unroll
    for (int r = 0; r < 8; ++r) av[r] = alpha[r];
#pragma unroll
    for (int ct = 0; ct < 16; ++ct) o[ct] = o[ct] * av;

    // O += P*V: A = P (16x32), B from transposed V tile (contiguous nk).
    v16bf ap = load_a_frag_rm(&lP[wid][0][0], 32);
#pragma unroll
    for (int ct = 0; ct < 16; ++ct) {
      v16bf bv = load_b_frag_nk(&lVT[cur][ct * 16][0], 32);
      o[ct] = wmma_bf16(ap, bv, o[ct]);
    }
    __syncthreads();
  }

  // Epilogue: normalize, stage through (now free) lK, coalesced b128 stores.
  __bf16* stage = (wid < 2) ? &lK[0][wid * 16][0] : &lK[1][(wid - 2) * 16][0];
  v8f iv;
#pragma unroll
  for (int r = 0; r < 8; ++r) iv[r] = 1.0f / rsum[r];
#pragma unroll
  for (int ct = 0; ct < 16; ++ct) {
    v8f on = o[ct] * iv;
#pragma unroll
    for (int r = 0; r < 8; ++r)
      stage[(size_t)(r + 8 * h) * CDIM + ct * 16 + col] = (__bf16)on[r];
  }
  const uint4* sst = (const uint4*)stage;
  uint4* gout = (uint4*)(xr + ((size_t)b * NDIM + n0) * CDIM);
#pragma unroll
  for (int i = 0; i < 16; ++i) {
    int l = lane + i * 32;   // 512 uint4 = 16 rows x 256 bf16
    gout[l] = sst[l];
  }
}

// ---------------------------------------------------------------------------
// Kernel 3: d = x^T - (x_r @ w_out^T + b_out)   (bf16 [B,N,C])
// ---------------------------------------------------------------------------
__global__ void __launch_bounds__(128)
proj_sub_kernel(const __bf16* __restrict__ xr, const float* __restrict__ w,
                const float* __restrict__ bias, const float* __restrict__ x,
                __bf16* __restrict__ d) {
  __shared__ alignas(16) __bf16 lA[2][64][32];
  __shared__ alignas(16) __bf16 lB[2][64][32];
  int b = blockIdx.z, n0 = blockIdx.x * 64, o0 = blockIdx.y * 64;
  int tid = threadIdx.x, lane = tid & 31, wid = tid >> 5;
  int wm = (wid & 1) * 32, wn = (wid >> 1) * 32;

  auto stage = [&](int buf, int kc) {
    uint4* sa = (uint4*)&lA[buf][0][0];
#pragma unroll
    for (int i = 0; i < 2; ++i) {
      int l = tid + i * 128;
      int ml = l >> 2, part = l & 3;
      sa[l] = *(const uint4*)(xr + ((size_t)b * NDIM + n0 + ml) * CDIM + kc +
                              part * 8);
    }
#pragma unroll
    for (int i = 0; i < 4; ++i) {
      int l = tid + i * 128;
      int ol = l >> 3, kp = l & 7;
      float4 wv = *(const float4*)(w + (size_t)(o0 + ol) * CDIM + kc + kp * 4);
      bf16x4 pk;
      pk.b[0] = (__bf16)wv.x; pk.b[1] = (__bf16)wv.y;
      pk.b[2] = (__bf16)wv.z; pk.b[3] = (__bf16)wv.w;
      *(uint2*)&lB[buf][ol][kp * 4] = pk.u2;
    }
  };

  v8f acc[2][2];
#pragma unroll
  for (int i = 0; i < 2; ++i)
#pragma unroll
    for (int j = 0; j < 2; ++j) acc[i][j] = v8f_zero();

  stage(0, 0);
  __syncthreads();
#pragma unroll
  for (int step = 0; step < 8; ++step) {
    int cur = step & 1;
    if (step < 7) stage(cur ^ 1, (step + 1) * 32);
#pragma unroll
    for (int i = 0; i < 2; ++i) {
      v16bf a = load_a_frag_rm(&lA[cur][wm + i * 16][0], 32);
#pragma unroll
      for (int j = 0; j < 2; ++j) {
        v16bf bb = load_b_frag_nk(&lB[cur][wn + j * 16][0], 32);
        acc[i][j] = wmma_bf16(a, bb, acc[i][j]);
      }
    }
    __syncthreads();
  }

  int h = lane >> 4, col = lane & 15;
#pragma unroll
  for (int i = 0; i < 2; ++i)
#pragma unroll
    for (int j = 0; j < 2; ++j) {
      int oc = o0 + wn + j * 16 + col;
      float bv = bias[oc];
#pragma unroll
      for (int r = 0; r < 8; ++r) {
        int nr = n0 + wm + i * 16 + r + 8 * h;
        float xt = x[((size_t)b * CDIM + oc) * NDIM + nr];
        d[((size_t)b * NDIM + nr) * CDIM + oc] =
            (__bf16)(xt - (acc[i][j][r] + bv));
      }
    }
}

// ---------------------------------------------------------------------------
// Kernel 4: y = d @ w_conv^T + b_conv (f32 [B,N,C]) + per-channel sum/sumsq.
// ---------------------------------------------------------------------------
__global__ void __launch_bounds__(128)
conv_stats_kernel(const __bf16* __restrict__ dmat, const float* __restrict__ w,
                  const float* __restrict__ bias, float* __restrict__ y,
                  float* __restrict__ stats) {
  __shared__ alignas(16) __bf16 lA[2][64][32];
  __shared__ alignas(16) __bf16 lB[2][64][32];
  int b = blockIdx.z, n0 = blockIdx.x * 64, o0 = blockIdx.y * 64;
  int tid = threadIdx.x, lane = tid & 31, wid = tid >> 5;
  int wm = (wid & 1) * 32, wn = (wid >> 1) * 32;

  auto stage = [&](int buf, int kc) {
    uint4* sa = (uint4*)&lA[buf][0][0];
#pragma unroll
    for (int i = 0; i < 2; ++i) {
      int l = tid + i * 128;
      int ml = l >> 2, part = l & 3;
      sa[l] = *(const uint4*)(dmat + ((size_t)b * NDIM + n0 + ml) * CDIM + kc +
                              part * 8);
    }
#pragma unroll
    for (int i = 0; i < 4; ++i) {
      int l = tid + i * 128;
      int ol = l >> 3, kp = l & 7;
      float4 wv = *(const float4*)(w + (size_t)(o0 + ol) * CDIM + kc + kp * 4);
      bf16x4 pk;
      pk.b[0] = (__bf16)wv.x; pk.b[1] = (__bf16)wv.y;
      pk.b[2] = (__bf16)wv.z; pk.b[3] = (__bf16)wv.w;
      *(uint2*)&lB[buf][ol][kp * 4] = pk.u2;
    }
  };

  v8f acc[2][2];
#pragma unroll
  for (int i = 0; i < 2; ++i)
#pragma unroll
    for (int j = 0; j < 2; ++j) acc[i][j] = v8f_zero();

  stage(0, 0);
  __syncthreads();
#pragma unroll
  for (int step = 0; step < 8; ++step) {
    int cur = step & 1;
    if (step < 7) stage(cur ^ 1, (step + 1) * 32);
#pragma unroll
    for (int i = 0; i < 2; ++i) {
      v16bf a = load_a_frag_rm(&lA[cur][wm + i * 16][0], 32);
#pragma unroll
      for (int j = 0; j < 2; ++j) {
        v16bf bb = load_b_frag_nk(&lB[cur][wn + j * 16][0], 32);
        acc[i][j] = wmma_bf16(a, bb, acc[i][j]);
      }
    }
    __syncthreads();
  }

  int h = lane >> 4, col = lane & 15;
#pragma unroll
  for (int j = 0; j < 2; ++j) {
    int oc = o0 + wn + j * 16 + col;
    float bv = bias[oc];
    float csum = 0.0f, csq = 0.0f;
#pragma unroll
    for (int i = 0; i < 2; ++i)
#pragma unroll
      for (int r = 0; r < 8; ++r) {
        int nr = n0 + wm + i * 16 + r + 8 * h;
        float yv = acc[i][j][r] + bv;
        y[((size_t)b * NDIM + nr) * CDIM + oc] = yv;
        csum += yv;
        csq += yv * yv;
      }
    csum += __shfl_xor(csum, 16, 32);
    csq  += __shfl_xor(csq, 16, 32);
    if (h == 0) {
      atomicAdd(&stats[oc], csum);
      atomicAdd(&stats[256 + oc], csq);
    }
  }
}

// ---------------------------------------------------------------------------
// Small kernels: stats zero / BN finalize / fused BN+ReLU+residual epilogue.
// ---------------------------------------------------------------------------
__global__ void zero_stats_kernel(float* __restrict__ s) {
  if (threadIdx.x < 512) s[threadIdx.x] = 0.0f;
}

__global__ void bn_finalize_kernel(float* __restrict__ s,
                                   const float* __restrict__ gamma,
                                   const float* __restrict__ beta) {
  int c = threadIdx.x;
  if (c < CDIM) {
    const float inv = 1.0f / (float)(BDIM * NDIM);
    float mean = s[c] * inv;
    float var = s[256 + c] * inv - mean * mean;
    float sc = gamma[c] * rsqrtf(var + 1e-5f);
    s[512 + c] = sc;
    s[768 + c] = beta[c] - mean * sc;
  }
}

__global__ void __launch_bounds__(1024)
epilogue_kernel(const float* __restrict__ y, const float* __restrict__ x,
                const float* __restrict__ stats, float* __restrict__ out) {
  __shared__ float t[32][33];
  int b = blockIdx.z, n0 = blockIdx.x * 32, c0 = blockIdx.y * 32;
  int tx = threadIdx.x, ty = threadIdx.y;
  t[ty][tx] = y[((size_t)b * NDIM + n0 + ty) * CDIM + c0 + tx];
  __syncthreads();
  int c = c0 + ty, n = n0 + tx;
  float yv = t[tx][ty] * stats[512 + c] + stats[768 + c];
  size_t idx = ((size_t)b * CDIM + c) * NDIM + n;
  out[idx] = x[idx] + fmaxf(yv, 0.0f);
}

// ---------------------------------------------------------------------------
extern "C" void kernel_launch(void* const* d_in, const int* in_sizes, int n_in,
                              void* d_out, int out_size, void* d_ws,
                              size_t ws_size, hipStream_t stream) {
  const float* x      = (const float*)d_in[0];
  const float* w_qkv  = (const float*)d_in[1];
  const float* w_out  = (const float*)d_in[2];
  const float* b_out  = (const float*)d_in[3];
  const float* w_conv = (const float*)d_in[4];
  const float* b_conv = (const float*)d_in[5];
  const float* gamma  = (const float*)d_in[6];
  const float* beta   = (const float*)d_in[7];
  float* out = (float*)d_out;

  // Workspace layout (bytes); dead q/k/v regions are reused.
  char* ws = (char*)d_ws;
  const size_t SEG = (size_t)BDIM * NDIM * CDIM * 2;  // 16 MiB bf16 tensor
  __bf16* q    = (__bf16*)(ws);
  __bf16* kbuf = (__bf16*)(ws + SEG);
  __bf16* vtb  = (__bf16*)(ws + 2 * SEG);   // V transposed [B,C,N]
  __bf16* xrb  = (__bf16*)(ws + 3 * SEG);
  float*  stats = (float*)(ws + 4 * SEG);   // 1024 floats
  __bf16* dbuf = q;                         // reuse (q dead after attention)
  float*  ybuf = (float*)(ws + SEG);        // reuse k+v

  qkv_kernel<<<dim3(NDIM / 64, (3 * CDIM) / 64, BDIM), 128, 0, stream>>>(
      x, w_qkv, q, kbuf, vtb);
  attn_kernel<<<dim3(NDIM / 64, BDIM), 128, 0, stream>>>(q, kbuf, vtb, xrb);
  proj_sub_kernel<<<dim3(NDIM / 64, CDIM / 64, BDIM), 128, 0, stream>>>(
      xrb, w_out, b_out, x, dbuf);
  zero_stats_kernel<<<1, 512, 0, stream>>>(stats);
  conv_stats_kernel<<<dim3(NDIM / 64, CDIM / 64, BDIM), 128, 0, stream>>>(
      dbuf, w_conv, b_conv, ybuf, stats);
  bn_finalize_kernel<<<1, 256, 0, stream>>>(stats, gamma, beta);
  epilogue_kernel<<<dim3(NDIM / 32, CDIM / 32, BDIM), dim3(32, 32), 0,
                    stream>>>(ybuf, x, stats, out);
}
